// StructureModule_18159121727870
// MI455X (gfx1250) — compile-verified, matrix operand order checked by reading, not run on hardware
//
#include <hip/hip_runtime.h>
#include <hip/hip_bf16.h>

typedef __bf16 bf16_t;
typedef __attribute__((ext_vector_type(16))) __bf16 v16bf;
typedef __attribute__((ext_vector_type(8)))  float  v8f;
typedef unsigned int u32x4 __attribute__((ext_vector_type(4)));
typedef int          i32x8 __attribute__((ext_vector_type(8)));
typedef int          i32x4 __attribute__((ext_vector_type(4)));

#define GF_BIAS  1
#define GF_RELU  2
#define GF_ACCUM 4

// ---------------------------------------------------------------------------
// Plain batched GEMM (operands streamed from global):
//   C[b] = A[b](MxK, bf16) * W[b](NcxK bf16, pre-transposed) (+bias)(+relu)
// One wave per 16x16 tile, K in steps of 32 via v_wmma_f32_16x16x32_bf16.
// ---------------------------------------------------------------------------
__global__ __launch_bounds__(32)
void k_gemm_bf16(const bf16_t* __restrict__ A, int lda, long sA,
                 const bf16_t* __restrict__ Wt, int ldw, long sW,
                 const float* __restrict__ bias,
                 float* __restrict__ C, int ldc, long sC,
                 bf16_t* __restrict__ Cb, int ldcb, long sCb,
                 int K, int flags)
{
  const int nt = blockIdx.x, mt = blockIdx.y, b = blockIdx.z;
  const int lane = threadIdx.x;
  const int half = lane >> 4;
  const int r15  = lane & 15;
  const bf16_t* arow = A  + (long)b*sA + ((long)mt*16 + r15) * (long)lda;
  const bf16_t* wrow = Wt + (long)b*sW + ((long)nt*16 + r15) * (long)ldw;

  v8f acc = {0.f,0.f,0.f,0.f,0.f,0.f,0.f,0.f};
  if (flags & GF_ACCUM) {
    const float* Ci = C + (long)b*sC + ((long)mt*16)*(long)ldc + (long)nt*16;
#pragma unroll
    for (int r=0;r<8;r++) acc[r] = Ci[(long)(r + 8*half)*ldc + r15];
  }
  for (int k0=0;k0<K;k0+=32){
    v16bf av, bv;
#pragma unroll
    for (int g=0; g<2; ++g){
      const int kb = k0 + g*16 + half*8;
      const unsigned int* ap = (const unsigned int*)(arow + kb);
      const unsigned int* wp = (const unsigned int*)(wrow + kb);
#pragma unroll
      for (int d=0; d<4; ++d){
        ((unsigned int*)&av)[g*4+d] = ap[d];
        ((unsigned int*)&bv)[g*4+d] = wp[d];
      }
    }
    acc = __builtin_amdgcn_wmma_f32_16x16x32_bf16(false, av, false, bv,
                                                  (short)0, acc, false, false);
  }
  const float bc = (flags & GF_BIAS) ? bias[(long)nt*16 + r15] : 0.f;
#pragma unroll
  for (int r=0;r<8;r++){
    float v = acc[r] + bc;
    if (flags & GF_RELU) v = v > 0.f ? v : 0.f;
    const long row = (long)mt*16 + r + 8*half;
    const long col = (long)nt*16 + r15;
    if (C)  C [(long)b*sC  + row*(long)ldc  + col] = v;
    if (Cb) Cb[(long)b*sCb + row*(long)ldcb + col] = (bf16_t)v;
  }
}

// ---------------------------------------------------------------------------
// TDM 2D tile load: global (tile_k x tile_rows, bf16, row stride in elements)
// -> LDS (contiguous). D# per CDNA5 ISA.
//   group0: count=1 | lds_addr[63:32] | global_addr[120:64] | type=2[127:126]
//   group1: data_size=1(2B)[17:16], tensor_dim0=tile_k[79:48],
//           tensor_dim1=tile_rows[111:80], tile_dim0[127:112],
//           tile_dim1[143:128], tensor_dim0_stride[207:160]
// ---------------------------------------------------------------------------
__device__ __forceinline__ void tdm_load_2d(unsigned lds_off, const void* gp,
                                            unsigned tile_k, unsigned tile_rows,
                                            unsigned row_stride){
  unsigned long long ga = (unsigned long long)gp;
  u32x4 g0;
  g0.x = 1u;                                           // count=1 (valid user D#)
  g0.y = lds_off;                                      // lds_addr
  g0.z = (unsigned)(ga & 0xffffffffu);                 // global_addr[31:0]
  g0.w = (unsigned)((ga >> 32) & 0x1ffffffu) | (2u << 30); // addr[56:32] | type=2
  i32x8 g1;
  g1[0] = (int)(1u << 16);                             // data_size=1 -> 2 bytes
  g1[1] = (int)((tile_k & 0xffffu) << 16);             // tensor_dim0 lo16
  g1[2] = (int)(((tile_k >> 16) & 0xffffu) | ((tile_rows & 0xffffu) << 16));
  g1[3] = (int)(((tile_rows >> 16) & 0xffffu) | ((tile_k & 0xffffu) << 16)); // tile_dim0
  g1[4] = (int)(tile_rows & 0xffffu);                  // tile_dim1 (tile_dim2=0)
  g1[5] = (int)row_stride;                             // tensor_dim0_stride lo32
  g1[6] = 0;
  g1[7] = 0;
  i32x4 gz = {0,0,0,0};
#if defined(__clang_major__) && (__clang_major__ >= 23)
  i32x8 gz8 = {0,0,0,0,0,0,0,0};
  __builtin_amdgcn_tensor_load_to_lds(g0, g1, gz, gz, gz8, 0);
#else
  __builtin_amdgcn_tensor_load_to_lds(g0, g1, gz, gz, 0);
#endif
}

// ---------------------------------------------------------------------------
// LDS-staged GEMM for fat tiles: block = 4 waves -> 32x32 macro-tile.
// A/W K-chunks (32x64 bf16) staged to LDS by TDM, double-buffered,
// fenced with s_wait_tensorcnt + workgroup barrier. Each LDS byte feeds 2 waves.
// Requires M%32==0, Nc%32==0, K%64==0.
// ---------------------------------------------------------------------------
#define KC 64
__global__ __launch_bounds__(128)
void k_gemm_tdm(const bf16_t* __restrict__ A, int lda, long sA,
                const bf16_t* __restrict__ Wt, int ldw, long sW,
                const float* __restrict__ bias,
                float* __restrict__ C, int ldc, long sC,
                bf16_t* __restrict__ Cb, int ldcb, long sCb,
                int K, int flags)
{
  __shared__ __align__(16) bf16_t Ab[2][32*KC];
  __shared__ __align__(16) bf16_t Bb[2][32*KC];
  const int tid  = threadIdx.x;
  const int wave = tid >> 5;
  const int lane = tid & 31;
  const int half = lane >> 4, r15 = lane & 15;
  const int mi = wave >> 1, ni = wave & 1;
  const int m0 = blockIdx.y*32, n0 = blockIdx.x*32, b = blockIdx.z;
  const bf16_t* Ag = A  + (long)b*sA + (long)m0*lda;
  const bf16_t* Wg = Wt + (long)b*sW + (long)n0*ldw;
  const unsigned ldsA0 = (unsigned)(unsigned long long)&Ab[0][0];
  const unsigned ldsA1 = (unsigned)(unsigned long long)&Ab[1][0];
  const unsigned ldsB0 = (unsigned)(unsigned long long)&Bb[0][0];
  const unsigned ldsB1 = (unsigned)(unsigned long long)&Bb[1][0];
  const int nch = K / KC;

  if (wave == 0){
    tdm_load_2d(ldsA0, Ag, KC, 32, (unsigned)lda);
    tdm_load_2d(ldsB0, Wg, KC, 32, (unsigned)ldw);
  }
  v8f acc = {0.f,0.f,0.f,0.f,0.f,0.f,0.f,0.f};
  if (flags & GF_ACCUM) {
    const float* Ci = C + (long)b*sC + ((long)m0 + mi*16)*(long)ldc + n0 + ni*16;
#pragma unroll
    for (int r=0;r<8;r++) acc[r] = Ci[(long)(r + 8*half)*ldc + r15];
  }
  for (int c=0; c<nch; ++c){
    const int buf = c & 1;
    if (wave == 0){
      if (c+1 < nch){
        tdm_load_2d(buf ? ldsA0 : ldsA1, Ag + (long)(c+1)*KC, KC, 32, (unsigned)lda);
        tdm_load_2d(buf ? ldsB0 : ldsB1, Wg + (long)(c+1)*KC, KC, 32, (unsigned)ldw);
        __builtin_amdgcn_s_wait_tensorcnt(2);   // chunk c complete, c+1 in flight
      } else {
        __builtin_amdgcn_s_wait_tensorcnt(0);
      }
    }
    __syncthreads();
    const bf16_t* ar = &Ab[buf][(mi*16 + r15)*KC];
    const bf16_t* br = &Bb[buf][(ni*16 + r15)*KC];
#pragma unroll
    for (int k0=0;k0<KC;k0+=32){
      v16bf av, bv;
#pragma unroll
      for (int g=0; g<2; ++g){
        const int kb = k0 + g*16 + half*8;
        const uint4* ap = (const uint4*)(ar + kb);   // ds_load_b128
        const uint4* bp = (const uint4*)(br + kb);
        uint4 a4 = *ap, b4 = *bp;
        ((unsigned int*)&av)[g*4+0]=a4.x; ((unsigned int*)&av)[g*4+1]=a4.y;
        ((unsigned int*)&av)[g*4+2]=a4.z; ((unsigned int*)&av)[g*4+3]=a4.w;
        ((unsigned int*)&bv)[g*4+0]=b4.x; ((unsigned int*)&bv)[g*4+1]=b4.y;
        ((unsigned int*)&bv)[g*4+2]=b4.z; ((unsigned int*)&bv)[g*4+3]=b4.w;
      }
      acc = __builtin_amdgcn_wmma_f32_16x16x32_bf16(false, av, false, bv,
                                                    (short)0, acc, false, false);
    }
    __syncthreads();
  }
  const float bc = (flags & GF_BIAS) ? bias[(long)n0 + ni*16 + r15] : 0.f;
#pragma unroll
  for (int r=0;r<8;r++){
    float v = acc[r] + bc;
    if (flags & GF_RELU) v = v > 0.f ? v : 0.f;
    const long row = (long)m0 + mi*16 + r + 8*half;
    const long col = (long)n0 + ni*16 + r15;
    if (C)  C [(long)b*sC  + row*(long)ldc  + col] = v;
    if (Cb) Cb[(long)b*sCb + row*(long)ldcb + col] = (bf16_t)v;
  }
}

// W (din x dout, f32 row-major) -> Wt (dout x din, bf16)
__global__ void k_transpose_w(const float* __restrict__ W, bf16_t* __restrict__ Wt,
                              int din, int dout){
  int idx = blockIdx.x*256 + threadIdx.x;
  if (idx >= din*dout) return;
  int n = idx / din, k = idx % din;
  Wt[idx] = (bf16_t)W[(long)k*dout + n];
}

__global__ void k_f2bf(const float* __restrict__ x, bf16_t* __restrict__ y, long n, int relu){
  long i = (long)blockIdx.x*256 + threadIdx.x;
  if (i >= n) return;
  float v = x[i];
  if (relu) v = v > 0.f ? v : 0.f;
  y[i] = (bf16_t)v;
}

__global__ void k_copy(const float* __restrict__ src, float* __restrict__ dst, long n){
  long i = (long)blockIdx.x*256 + threadIdx.x;
  if (i < n) dst[i] = src[i];
}

// layernorm over D (<=384) per row; optional residual add; optional f32/bf16/relu-bf16 outputs
__global__ __launch_bounds__(128)
void k_layernorm(const float* __restrict__ x, const float* __restrict__ res,
                 const float* __restrict__ g, const float* __restrict__ bt, int D,
                 float* of32, bf16_t* obf, bf16_t* orelu){
  const int row = blockIdx.x, tid = threadIdx.x;
  x += (long)row*D; if (res) res += (long)row*D;
  __shared__ float vals[384];
  __shared__ float red[128];
  float lsum = 0.f;
  for (int i=tid;i<D;i+=128){ float v = x[i] + (res?res[i]:0.f); vals[i]=v; lsum+=v; }
  red[tid]=lsum; __syncthreads();
  for (int s=64;s>0;s>>=1){ if(tid<s) red[tid]+=red[tid+s]; __syncthreads(); }
  float mean = red[0]/(float)D; __syncthreads();
  float lvar=0.f; for(int i=tid;i<D;i+=128){ float d=vals[i]-mean; lvar+=d*d; }
  red[tid]=lvar; __syncthreads();
  for (int s=64;s>0;s>>=1){ if(tid<s) red[tid]+=red[tid+s]; __syncthreads(); }
  float inv = rsqrtf(red[0]/(float)D + 1e-5f);
  for(int i=tid;i<D;i+=128){
    float v = (vals[i]-mean)*inv*g[i]+bt[i];
    if(of32)  of32 [(long)row*D+i]=v;
    if(obf)   obf  [(long)row*D+i]=(bf16_t)v;
    if(orelu) orelu[(long)row*D+i]=(bf16_t)(v>0.f?v:0.f);
  }
}

// layernorm z rows (D=128), write bf16 TRANSPOSED: zt[i][c][j]
__global__ __launch_bounds__(128)
void k_ln_z(const float* __restrict__ z, const float* __restrict__ g,
            const float* __restrict__ bt, bf16_t* __restrict__ zt){
  const long row = blockIdx.x;       // i*512 + j
  const int tid = threadIdx.x;
  const int i = (int)(row >> 9), j = (int)(row & 511);
  float v = z[row*128 + tid];
  __shared__ float red[128];
  red[tid]=v; __syncthreads();
  for(int s=64;s>0;s>>=1){ if(tid<s) red[tid]+=red[tid+s]; __syncthreads(); }
  float mean = red[0]*(1.f/128.f); __syncthreads();
  float d=v-mean; red[tid]=d*d; __syncthreads();
  for(int s=64;s>0;s>>=1){ if(tid<s) red[tid]+=red[tid+s]; __syncthreads(); }
  float inv = rsqrtf(red[0]*(1.f/128.f)+1e-5f);
  zt[((long)i*128+tid)*512 + j] = (bf16_t)((v-mean)*inv*g[tid]+bt[tid]);
}

// battn[h][i][j] = bias_b[h] + sum_c zt[i][c][j] * wb[c][h]   (loop invariant -> once)
__global__ __launch_bounds__(128)
void k_battn(const bf16_t* __restrict__ zt, const float* __restrict__ wb,
             const float* __restrict__ bb, float* __restrict__ battn){
  long idx = (long)blockIdx.x*128 + threadIdx.x;
  if (idx >= (long)512*512) return;
  int i = (int)(idx>>9), j = (int)(idx&511);
  const bf16_t* zrow = zt + (long)i*128*512 + j;
  float acc[12];
#pragma unroll
  for (int h=0;h<12;h++) acc[h]=bb[h];
  for (int c=0;c<128;c++){
    float zc = (float)zrow[(long)c*512];
#pragma unroll
    for (int h=0;h<12;h++) acc[h] += zc*wb[c*12+h];
  }
#pragma unroll
  for (int h=0;h<12;h++) battn[((long)h*512+i)*512+j]=acc[h];
}

__global__ void k_init_pose(float* quats, float* trans, float* rot){
  int i = blockIdx.x*128 + threadIdx.x; if (i>=512) return;
  quats[i*4+0]=1.f; quats[i*4+1]=0.f; quats[i*4+2]=0.f; quats[i*4+3]=0.f;
  trans[i*3+0]=0.f; trans[i*3+1]=0.f; trans[i*3+2]=0.f;
  for(int k=0;k<9;k++) rot[i*9+k] = (k%4==0)?1.f:0.f;
}

// Transform q/kv points to global frame; build vt (bf16 [h][16][512]) and vpt (bf16 [h][32][512], rows 24..31 zero)
__global__ __launch_bounds__(128)
void k_points(const float* __restrict__ qp_raw, const float* __restrict__ kvp_raw,
              const float* __restrict__ kv, const float* __restrict__ rot,
              const float* __restrict__ trans,
              float* __restrict__ qpg, float* __restrict__ kpg,
              bf16_t* __restrict__ vpt, bf16_t* __restrict__ vt){
  const int i = blockIdx.x; const int tid = threadIdx.x;
  __shared__ float R[9], T[3];
  if (tid<9) R[tid]=rot[i*9+tid];
  if (tid<3) T[tid]=trans[i*3+tid];
  __syncthreads();
  for (int t=tid; t<480; t+=128){
    if (t < 48){                                   // q points: h*4+p
      const float* v = qp_raw + (long)i*144 + t*3;
      float x=v[0],y=v[1],z=v[2];
      float gx=R[0]*x+R[1]*y+R[2]*z+T[0];
      float gy=R[3]*x+R[4]*y+R[5]*z+T[1];
      float gz=R[6]*x+R[7]*y+R[8]*z+T[2];
      float* o = qpg + (long)i*144 + t*3; o[0]=gx;o[1]=gy;o[2]=gz;
    } else if (t < 192){                           // kv points: h*12+p
      int idx=t-48; int h=idx/12, p=idx%12;
      const float* v = kvp_raw + (long)i*432 + idx*3;
      float x=v[0],y=v[1],z=v[2];
      float gx=R[0]*x+R[1]*y+R[2]*z+T[0];
      float gy=R[3]*x+R[4]*y+R[5]*z+T[1];
      float gz=R[6]*x+R[7]*y+R[8]*z+T[2];
      if (p < 4){
        float* o = kpg + (((long)i*12+h)*4+p)*3; o[0]=gx;o[1]=gy;o[2]=gz;
      } else {
        long base = ((long)h*32 + (p-4)*3)*512 + i;
        vpt[base]=(bf16_t)gx; vpt[base+512]=(bf16_t)gy; vpt[base+1024]=(bf16_t)gz;
      }
    } else if (t < 288){                           // zero-pad vpt rows 24..31
      int idx=t-192; int h=idx/8, r=idx%8;
      vpt[((long)h*32+24+r)*512+i]=(bf16_t)0.f;
    } else {                                       // vt[h][c][j] = v part of kv
      int idx=t-288; int h=idx/16, c=idx%16;
      vt[((long)h*16+c)*512+i]=(bf16_t)kv[(long)i*384 + h*32+16+c];
    }
  }
}

// attention logits + softmax -> probs bf16 [i][16][512] (rows 12..15 zeroed)
__global__ __launch_bounds__(256)
void k_attn(const float* __restrict__ q, const float* __restrict__ kv,
            const float* __restrict__ qpg, const float* __restrict__ kpg,
            const float* __restrict__ battn, const float* __restrict__ mask,
            const float* __restrict__ head_w, bf16_t* __restrict__ probs){
  const int h = blockIdx.x, i = blockIdx.y, tid = threadIdx.x;
  bf16_t* prow = probs + ((long)i*16 + h)*512;
  if (h >= 12){ for(int j=tid;j<512;j+=256) prow[j]=(bf16_t)0.f; return; }
  __shared__ float qv[16], qp[12];
  __shared__ float red[256];
  if (tid<16) qv[tid] = q[(long)i*192 + h*16 + tid];
  if (tid<12) qp[tid] = qpg[(long)i*144 + h*12 + tid];
  __syncthreads();
  const float hw = log1pf(expf(head_w[h])) * 0.13608276f;   // softplus * sqrt(1/54)
  const float mi = mask[i];
  float lg[2];
#pragma unroll
  for (int r=0;r<2;r++){
    int j = tid + r*256;
    const float* krow = kv + (long)j*384 + h*32;
    float qk=0.f;
#pragma unroll
    for(int c=0;c<16;c++) qk += qv[c]*krow[c];
    const float* kp = kpg + ((long)j*12 + h)*4*3;
    float d2=0.f;
#pragma unroll
    for(int t=0;t<12;t++){ float d = qp[t]-kp[t]; d2 += d*d; }
    lg[r] = qk*0.14433757f + 0.57735027f*battn[((long)h*512+i)*512+j]
          - 0.5f*hw*d2 + 100000.0f*(mi*mask[j]-1.0f);
  }
  float m = fmaxf(lg[0],lg[1]); red[tid]=m; __syncthreads();
  for(int s=128;s>0;s>>=1){ if(tid<s) red[tid]=fmaxf(red[tid],red[tid+s]); __syncthreads(); }
  float mx = red[0]; __syncthreads();
  float e0=expf(lg[0]-mx), e1=expf(lg[1]-mx);
  red[tid]=e0+e1; __syncthreads();
  for(int s=128;s>0;s>>=1){ if(tid<s) red[tid]+=red[tid+s]; __syncthreads(); }
  float inv = 1.0f/red[0];
  prow[tid]     = (bf16_t)(e0*inv);
  prow[tid+256] = (bf16_t)(e1*inv);
}

// assemble feats bf16 [i][2112] = [o(192) | x(96) | y(96) | z(96) | norm(96) | o_pair(1536)]
__global__ __launch_bounds__(256)
void k_feats(const float* __restrict__ ob, const float* __restrict__ optb,
             const float* __restrict__ opair, const float* __restrict__ rot,
             const float* __restrict__ trans, bf16_t* __restrict__ feats){
  const int i = blockIdx.x, tid = threadIdx.x;
  __shared__ float R[9], T[3];
  if(tid<9)R[tid]=rot[i*9+tid];
  if(tid<3)T[tid]=trans[i*3+tid];
  __syncthreads();
  bf16_t* f = feats + (long)i*2112;
  for (int c=tid;c<192;c+=256){ int h=c>>4, cc=c&15; f[c]=(bf16_t)ob[((long)h*512+i)*16+cc]; }
  for (int t=tid;t<96;t+=256){
    int h=t>>3, p=t&7;
    const float* v = optb + ((long)h*512+i)*32 + p*3;
    float x=v[0]-T[0], y=v[1]-T[1], z=v[2]-T[2];
    float lx = R[0]*x+R[3]*y+R[6]*z;        // rot^T
    float ly = R[1]*x+R[4]*y+R[7]*z;
    float lz = R[2]*x+R[5]*y+R[8]*z;
    float nr = sqrtf(lx*lx+ly*ly+lz*lz+1e-8f);
    f[192+t]=(bf16_t)lx; f[288+t]=(bf16_t)ly; f[384+t]=(bf16_t)lz; f[480+t]=(bf16_t)nr;
  }
  for (int c=tid;c<1536;c+=256) f[576+c]=(bf16_t)opair[(long)i*2048+c];
}

// backbone update: upd = s@W+b; quat_mul; trans += rot@upd[3:6]; normalize; new rot; frames7 out
__global__ __launch_bounds__(128)
void k_bbupdate(const float* __restrict__ s, const float* __restrict__ w,
                const float* __restrict__ b, float* quats, float* trans,
                float* rot, float* frames7){
  int i = blockIdx.x*128+threadIdx.x; if(i>=512) return;
  const float* srow = s + (long)i*384;
  float u[6];
#pragma unroll
  for(int o=0;o<6;o++){ float a=b[o]; for(int c=0;c<384;c++) a+=srow[c]*w[c*6+o]; u[o]=a; }
  float q0=quats[i*4+0], q1=quats[i*4+1], q2=quats[i*4+2], q3=quats[i*4+3];
  float x2=u[0], y2=u[1], z2=u[2];
  float nw = q0      - q1*x2 - q2*y2 - q3*z2;
  float nx = q0*x2 + q1      + q2*z2 - q3*y2;
  float ny = q0*y2 - q1*z2 + q2      + q3*x2;
  float nz = q0*z2 + q1*y2 - q2*x2 + q3;
  const float* R = rot + (long)i*9;
  float tx = R[0]*u[3]+R[1]*u[4]+R[2]*u[5] + trans[i*3+0];
  float ty = R[3]*u[3]+R[4]*u[4]+R[5]*u[5] + trans[i*3+1];
  float tz = R[6]*u[3]+R[7]*u[4]+R[8]*u[5] + trans[i*3+2];
  float inv = rsqrtf(nw*nw+nx*nx+ny*ny+nz*nz);
  nw*=inv; nx*=inv; ny*=inv; nz*=inv;
  quats[i*4+0]=nw; quats[i*4+1]=nx; quats[i*4+2]=ny; quats[i*4+3]=nz;
  trans[i*3+0]=tx; trans[i*3+1]=ty; trans[i*3+2]=tz;
  float* Ro = rot + (long)i*9;
  Ro[0]=1.f-2.f*(ny*ny+nz*nz); Ro[1]=2.f*(nx*ny-nw*nz);     Ro[2]=2.f*(nx*nz+nw*ny);
  Ro[3]=2.f*(nx*ny+nw*nz);     Ro[4]=1.f-2.f*(nx*nx+nz*nz); Ro[5]=2.f*(ny*nz-nw*nx);
  Ro[6]=2.f*(nx*nz-nw*ny);     Ro[7]=2.f*(ny*nz+nw*nx);     Ro[8]=1.f-2.f*(nx*nx+ny*ny);
  float* fo = frames7 + (long)i*7;
  fo[0]=nw; fo[1]=nx; fo[2]=ny; fo[3]=nz; fo[4]=tx*10.f; fo[5]=ty*10.f; fo[6]=tz*10.f;
}

// angle head: lin_out(relu(a)) (128->14), norms, unnorm/ang outputs
__global__ __launch_bounds__(128)
void k_angle_out(const float* __restrict__ a, const float* __restrict__ w,
                 const float* __restrict__ b, float* unnorm_out, float* ang_out){
  int i = blockIdx.x*128+threadIdx.x; if(i>=512) return;
  const float* ar = a + (long)i*128;
  float o[14];
#pragma unroll
  for(int t=0;t<14;t++){
    float acc=b[t];
    for(int c=0;c<128;c++){ float v=ar[c]; v=v>0.f?v:0.f; acc+=v*w[c*14+t]; }
    o[t]=acc;
  }
#pragma unroll
  for(int p=0;p<7;p++){
    float s0=o[2*p], s1=o[2*p+1];
    float n = sqrtf(fmaxf(s0*s0+s1*s1, 1e-8f));
    unnorm_out[(long)i*14+2*p]=s0; unnorm_out[(long)i*14+2*p+1]=s1;
    ang_out[(long)i*14+2*p]=s0/n;  ang_out[(long)i*14+2*p+1]=s1/n;
  }
}

__device__ __forceinline__ void mm3(float* o, const float* A, const float* B){
#pragma unroll
  for(int r=0;r<3;r++)
#pragma unroll
    for(int c=0;c<3;c++)
      o[r*3+c]=A[r*3+0]*B[0*3+c]+A[r*3+1]*B[1*3+c]+A[r*3+2]*B[2*3+c];
}
__device__ __forceinline__ void mv3(float* o, const float* A, const float* v){
#pragma unroll
  for(int r=0;r<3;r++) o[r]=A[r*3+0]*v[0]+A[r*3+1]*v[1]+A[r*3+2]*v[2];
}

// torsion -> frames -> atom positions; side44 + poss outputs
__global__ __launch_bounds__(64)
void k_frames(const int* __restrict__ aatype, const float* __restrict__ df,
              const int* __restrict__ gidx, const float* __restrict__ amask,
              const float* __restrict__ lit, const float* __restrict__ quats,
              const float* __restrict__ trans, const float* __restrict__ ang,
              float* __restrict__ side44, float* __restrict__ poss){
  int i = blockIdx.x*64+threadIdx.x; if(i>=512) return;
  int aat = aatype[i];
  float qw=quats[i*4+0],qx=quats[i*4+1],qy=quats[i*4+2],qz=quats[i*4+3];
  float Rb[9];
  Rb[0]=1.f-2.f*(qy*qy+qz*qz); Rb[1]=2.f*(qx*qy-qw*qz);     Rb[2]=2.f*(qx*qz+qw*qy);
  Rb[3]=2.f*(qx*qy+qw*qz);     Rb[4]=1.f-2.f*(qx*qx+qz*qz); Rb[5]=2.f*(qy*qz-qw*qx);
  Rb[6]=2.f*(qx*qz-qw*qy);     Rb[7]=2.f*(qy*qz+qw*qx);     Rb[8]=1.f-2.f*(qx*qx+qy*qy);
  float T[3]={trans[i*3+0]*10.f, trans[i*3+1]*10.f, trans[i*3+2]*10.f};
  float Rf[8][9], tf[8][3];
#pragma unroll
  for (int f=0; f<8; f++){
    const float* d = df + ((long)aat*8 + f)*16;
    float sn = (f==0) ? 0.f : ang[(long)i*14+(f-1)*2];
    float cs = (f==0) ? 1.f : ang[(long)i*14+(f-1)*2+1];
#pragma unroll
    for(int r=0;r<3;r++){
      float d0=d[r*4+0], d1=d[r*4+1], d2=d[r*4+2];
      Rf[f][r*3+0]=d0;
      Rf[f][r*3+1]= d1*cs + d2*sn;
      Rf[f][r*3+2]=-d1*sn + d2*cs;
    }
    tf[f][0]=d[3]; tf[f][1]=d[7]; tf[f][2]=d[11];
  }
  float Rp[9], tp[3];
#pragma unroll
  for(int k=0;k<9;k++) Rp[k]=Rf[4][k];
#pragma unroll
  for(int k=0;k<3;k++) tp[k]=tf[4][k];
  for (int f=5; f<8; f++){
    float Rn[9], tn[3];
    mm3(Rn, Rp, Rf[f]); mv3(tn, Rp, tf[f]);
#pragma unroll
    for(int k=0;k<3;k++) tn[k]+=tp[k];
#pragma unroll
    for(int k=0;k<9;k++){ Rf[f][k]=Rn[k]; Rp[k]=Rn[k]; }
#pragma unroll
    for(int k=0;k<3;k++){ tf[f][k]=tn[k]; tp[k]=tn[k]; }
  }
  float Rg[8][9], tg[8][3];
  float* s44 = side44 + (long)i*128;
#pragma unroll
  for (int f=0; f<8; f++){
    mm3(Rg[f], Rb, Rf[f]); mv3(tg[f], Rb, tf[f]);
#pragma unroll
    for(int k=0;k<3;k++) tg[f][k]+=T[k];
#pragma unroll
    for(int r=0;r<3;r++){
#pragma unroll
      for(int c=0;c<3;c++) s44[f*16+r*4+c]=Rg[f][r*3+c];
      s44[f*16+r*4+3]=tg[f][r];
    }
    s44[f*16+12]=0.f; s44[f*16+13]=0.f; s44[f*16+14]=0.f; s44[f*16+15]=1.f;
  }
  for (int a=0;a<14;a++){
    int g = gidx[aat*14+a];
    const float* L = lit + ((long)aat*14+a)*3;
    float m = amask[aat*14+a];
    float p[3]; mv3(p, Rg[g], L);
    poss[(long)i*42+a*3+0]=(p[0]+tg[g][0])*m;
    poss[(long)i*42+a*3+1]=(p[1]+tg[g][1])*m;
    poss[(long)i*42+a*3+2]=(p[2]+tg[g][2])*m;
  }
}

// ---------------------------------------------------------------------------
// Host orchestration
// ---------------------------------------------------------------------------
enum {
  IN_S=0, IN_Z, IN_MASK, IN_AATYPE,
  P_AB0L1_B, P_AB0L1_W, P_AB0L2_B, P_AB0L2_W,
  P_AB1L1_B, P_AB1L1_W, P_AB1L2_B, P_AB1L2_W,
  P_AIN_B, P_AIN_W, P_AINIT_B, P_AINIT_W, P_AOUT_B, P_AOUT_W,
  P_BB_B, P_BB_W,
  P_HEADW, P_LB_B, P_LB_W, P_LKV_B, P_LKV_W, P_LKVP_B, P_LKVP_W,
  P_LOUT_B, P_LOUT_W, P_LQ_B, P_LQ_W, P_LQP_B, P_LQP_W,
  P_LIN_B, P_LIN_W,
  P_LNIPA_B, P_LNIPA_G, P_LNS_B, P_LNS_G, P_LNZ_B, P_LNZ_G,
  P_T1_B, P_T1_W, P_T2_B, P_T2_W, P_T3_B, P_T3_W, P_TLN_B, P_TLN_G,
  IN_DF, IN_GIDX, IN_AMASK, IN_LIT
};

#define OFF_FRAMES7 0L
#define OFF_SIDE44  28672L
#define OFF_UNNORM  552960L
#define OFF_ANG     610304L
#define OFF_POS     667648L
#define OFF_STATES  839680L
#define OFF_SFINAL  2412544L

static inline void gemm(hipStream_t st, const bf16_t* A, int lda, long sA,
                        const bf16_t* Wt, int ldw, long sW, const float* bias,
                        float* C, int ldc, long sC, bf16_t* Cb, int ldcb, long sCb,
                        int M, int Nc, int K, int batch, int flags){
  if (M % 32 == 0 && Nc % 32 == 0 && K % 64 == 0){
    dim3 g(Nc/32, M/32, batch);
    k_gemm_tdm<<<g, 128, 0, st>>>(A,lda,sA,Wt,ldw,sW,bias,C,ldc,sC,Cb,ldcb,sCb,K,flags);
  } else {
    dim3 g(Nc/16, M/16, batch);
    k_gemm_bf16<<<g, 32, 0, st>>>(A,lda,sA,Wt,ldw,sW,bias,C,ldc,sC,Cb,ldcb,sCb,K,flags);
  }
}

extern "C" void kernel_launch(void* const* d_in, const int* in_sizes, int n_in,
                              void* d_out, int out_size, void* d_ws, size_t ws_size,
                              hipStream_t stream){
  (void)in_sizes; (void)n_in; (void)out_size; (void)ws_size;
  const float* s_in   = (const float*)d_in[IN_S];
  const float* z_in   = (const float*)d_in[IN_Z];
  const float* mask   = (const float*)d_in[IN_MASK];
  const int*   aatype = (const int*)  d_in[IN_AATYPE];
  const float* dframes= (const float*)d_in[IN_DF];
  const int*   gidx   = (const int*)  d_in[IN_GIDX];
  const float* amask  = (const float*)d_in[IN_AMASK];
  const float* lit    = (const float*)d_in[IN_LIT];
  float* out = (float*)d_out;
#define PF(i) ((const float*)d_in[i])

  // workspace bump allocator
  char* wsp = (char*)d_ws; size_t off = 0;
  auto alloc = [&](size_t bytes)->void*{
    void* p = wsp + off; off = (off + bytes + 255) & ~(size_t)255; return p; };

  bf16_t* zt     = (bf16_t*)alloc((size_t)512*512*128*2);
  float*  battn  = (float*) alloc((size_t)12*512*512*4);
  float*  s_cur  = (float*) alloc(512*384*4);
  bf16_t* s_bf   = (bf16_t*)alloc(512*384*2);
  bf16_t* s_relu = (bf16_t*)alloc(512*384*2);
  bf16_t* si_bf  = (bf16_t*)alloc(512*384*2);   // ln_s(s) bf16
  bf16_t* si_relu= (bf16_t*)alloc(512*384*2);
  bf16_t* Wt_lin = (bf16_t*)alloc(384*384*2);
  bf16_t* Wt_q   = (bf16_t*)alloc(384*192*2);
  bf16_t* Wt_kv  = (bf16_t*)alloc(384*384*2);
  bf16_t* Wt_qp  = (bf16_t*)alloc(384*144*2);
  bf16_t* Wt_kvp = (bf16_t*)alloc(384*432*2);
  bf16_t* Wt_out = (bf16_t*)alloc((size_t)2112*384*2);
  bf16_t* Wt_t1  = (bf16_t*)alloc(384*384*2);
  bf16_t* Wt_t2  = (bf16_t*)alloc(384*384*2);
  bf16_t* Wt_t3  = (bf16_t*)alloc(384*384*2);
  bf16_t* Wt_ain = (bf16_t*)alloc(384*128*2);
  bf16_t* Wt_aini= (bf16_t*)alloc(384*128*2);
  bf16_t* Wt_b0a = (bf16_t*)alloc(128*128*2);
  bf16_t* Wt_b0b = (bf16_t*)alloc(128*128*2);
  bf16_t* Wt_b1a = (bf16_t*)alloc(128*128*2);
  bf16_t* Wt_b1b = (bf16_t*)alloc(128*128*2);
  float*  q_f32  = (float*) alloc(512*192*4);
  float*  kv_f32 = (float*) alloc(512*384*4);
  float*  qp_raw = (float*) alloc(512*144*4);
  float*  kvp_raw= (float*) alloc(512*432*4);
  float*  qpg    = (float*) alloc(512*144*4);
  float*  kpg    = (float*) alloc(512*144*4);
  bf16_t* vt     = (bf16_t*)alloc((size_t)12*16*512*2);
  bf16_t* vpt    = (bf16_t*)alloc((size_t)12*32*512*2);
  bf16_t* probs  = (bf16_t*)alloc((size_t)512*16*512*2);
  float*  ob     = (float*) alloc((size_t)12*512*16*4);
  float*  optb   = (float*) alloc((size_t)12*512*32*4);
  float*  opair  = (float*) alloc((size_t)512*16*128*4);
  bf16_t* feats  = (bf16_t*)alloc((size_t)512*2112*2);
  float*  ipa_o  = (float*) alloc(512*384*4);
  bf16_t* t1bf   = (bf16_t*)alloc(512*384*2);
  bf16_t* t2bf   = (bf16_t*)alloc(512*384*2);
  float*  t3f    = (float*) alloc(512*384*4);
  float*  a_f32  = (float*) alloc(512*128*4);
  bf16_t* a_relu = (bf16_t*)alloc(512*128*2);
  bf16_t* t_abf  = (bf16_t*)alloc(512*128*2);
  float*  quats  = (float*) alloc(512*4*4);
  float*  trans  = (float*) alloc(512*3*4);
  float*  rot    = (float*) alloc(512*9*4);

  auto prep = [&](int wi, bf16_t* Wt, int din, int dout){
    int n = din*dout;
    k_transpose_w<<<(n+255)/256, 256, 0, stream>>>(PF(wi), Wt, din, dout);
  };
  // ---- loop-invariant preprocessing ----
  prep(P_LIN_W, Wt_lin, 384, 384);
  prep(P_LQ_W,  Wt_q,   384, 192);
  prep(P_LKV_W, Wt_kv,  384, 384);
  prep(P_LQP_W, Wt_qp,  384, 144);
  prep(P_LKVP_W,Wt_kvp, 384, 432);
  prep(P_LOUT_W,Wt_out, 2112,384);
  prep(P_T1_W,  Wt_t1,  384, 384);
  prep(P_T2_W,  Wt_t2,  384, 384);
  prep(P_T3_W,  Wt_t3,  384, 384);
  prep(P_AIN_W, Wt_ain, 384, 128);
  prep(P_AINIT_W,Wt_aini,384,128);
  prep(P_AB0L1_W,Wt_b0a,128, 128);
  prep(P_AB0L2_W,Wt_b0b,128, 128);
  prep(P_AB1L1_W,Wt_b1a,128, 128);
  prep(P_AB1L2_W,Wt_b1b,128, 128);

  // ln_s -> si_bf (for linear_in) + si_relu (for angle resnet)
  k_layernorm<<<512, 128, 0, stream>>>(s_in, nullptr, PF(P_LNS_G), PF(P_LNS_B), 384,
                                       nullptr, si_bf, si_relu);
  // ln_z -> zt (bf16 transposed, L2-resident); battn once (loop invariant)
  k_ln_z<<<512*512, 128, 0, stream>>>(z_in, PF(P_LNZ_G), PF(P_LNZ_B), zt);
  k_battn<<<(512*512+127)/128, 128, 0, stream>>>(zt, PF(P_LB_W), PF(P_LB_B), battn);
  k_init_pose<<<4, 128, 0, stream>>>(quats, trans, rot);
  // s = linear_in(ln_s(s))  -> s_cur f32 + s_bf bf16
  gemm(stream, si_bf,384,0, Wt_lin,384,0, PF(P_LIN_B),
       s_cur,384,0, s_bf,384,0, 512,384,384, 1, GF_BIAS);

  // ---- 8 structure blocks ----
  for (int blk=0; blk<8; ++blk){
    float* frames7_o = out + OFF_FRAMES7 + (long)blk*512*7;
    float* side44_o  = out + OFF_SIDE44  + (long)blk*512*128;
    float* unnorm_o  = out + OFF_UNNORM  + (long)blk*512*14;
    float* ang_o     = out + OFF_ANG     + (long)blk*512*14;
    float* pos_o     = out + OFF_POS     + (long)blk*512*42;
    float* states_o  = out + OFF_STATES  + (long)blk*512*384;

    // IPA projections (q/kv via TDM-staged path; point projections via plain path)
    gemm(stream, s_bf,384,0, Wt_q,  384,0, PF(P_LQ_B),  q_f32,  192,0, nullptr,0,0, 512,192,384,1, GF_BIAS);
    gemm(stream, s_bf,384,0, Wt_kv, 384,0, PF(P_LKV_B), kv_f32, 384,0, nullptr,0,0, 512,384,384,1, GF_BIAS);
    gemm(stream, s_bf,384,0, Wt_qp, 384,0, PF(P_LQP_B), qp_raw, 144,0, nullptr,0,0, 512,144,384,1, GF_BIAS);
    gemm(stream, s_bf,384,0, Wt_kvp,384,0, PF(P_LKVP_B),kvp_raw,432,0, nullptr,0,0, 512,432,384,1, GF_BIAS);
    k_points<<<512, 128, 0, stream>>>(qp_raw, kvp_raw, kv_f32, rot, trans, qpg, kpg, vpt, vt);
    // softmax probs (bf16, [i][16][512], rows 12..15 zero)
    { dim3 g(16, 512); k_attn<<<g, 256, 0, stream>>>(q_f32, kv_f32, qpg, kpg, battn,
                                                     mask, PF(P_HEADW), probs); }
    // o = probs @ v   (12 batched 512x512x16)
    gemm(stream, probs,16*512,512, vt, 512,16*512, nullptr,
         ob,16,(long)512*16, nullptr,0,0, 512,16,512, 12, 0);
    // o_pt = probs @ v_pts (12 batched 512x512x32, cols 24..31 zero) -> TDM path
    gemm(stream, probs,16*512,512, vpt,512,32*512, nullptr,
         optb,32,(long)512*32, nullptr,0,0, 512,32,512, 12, 0);
    // o_pair = probs_i @ z_i  (512 batched 16x512x128 against zt tiles)
    gemm(stream, probs,512,16*512, zt,512,(long)128*512, nullptr,
         opair,128,2048, nullptr,0,0, 16,128,512, 512, 0);
    k_feats<<<512, 256, 0, stream>>>(ob, optb, opair, rot, trans, feats);
    // linear_out (K = 2112) -> TDM path
    gemm(stream, feats,2112,0, Wt_out,2112,0, PF(P_LOUT_B),
         ipa_o,384,0, nullptr,0,0, 512,384,2112, 1, GF_BIAS);
    // s = LN(s + ipa_out)
    k_layernorm<<<512, 128, 0, stream>>>(ipa_o, s_cur, PF(P_LNIPA_G), PF(P_LNIPA_B), 384,
                                         s_cur, s_bf, nullptr);
    // transition
    gemm(stream, s_bf,384,0, Wt_t1,384,0, PF(P_T1_B), nullptr,0,0, t1bf,384,0, 512,384,384,1, GF_BIAS|GF_RELU);
    gemm(stream, t1bf,384,0, Wt_t2,384,0, PF(P_T2_B), nullptr,0,0, t2bf,384,0, 512,384,384,1, GF_BIAS|GF_RELU);
    gemm(stream, t2bf,384,0, Wt_t3,384,0, PF(P_T3_B), t3f,384,0, nullptr,0,0, 512,384,384,1, GF_BIAS);
    k_layernorm<<<512, 128, 0, stream>>>(t3f, s_cur, PF(P_TLN_G), PF(P_TLN_B), 384,
                                         s_cur, s_bf, s_relu);
    k_copy<<<(512*384+255)/256, 256, 0, stream>>>(s_cur, states_o, (long)512*384);
    // backbone update (uses old rot, writes new quats/trans/rot + frames7)
    k_bbupdate<<<4, 128, 0, stream>>>(s_cur, PF(P_BB_W), PF(P_BB_B), quats, trans, rot, frames7_o);
    // angle resnet
    gemm(stream, s_relu, 384,0, Wt_ain, 384,0, PF(P_AIN_B),  a_f32,128,0, nullptr,0,0, 512,128,384,1, GF_BIAS);
    gemm(stream, si_relu,384,0, Wt_aini,384,0, PF(P_AINIT_B),a_f32,128,0, nullptr,0,0, 512,128,384,1, GF_BIAS|GF_ACCUM);
    k_f2bf<<<(512*128+255)/256, 256, 0, stream>>>(a_f32, a_relu, (long)512*128, 1);
    gemm(stream, a_relu,128,0, Wt_b0a,128,0, PF(P_AB0L1_B), nullptr,0,0, t_abf,128,0, 512,128,128,1, GF_BIAS|GF_RELU);
    gemm(stream, t_abf, 128,0, Wt_b0b,128,0, PF(P_AB0L2_B), a_f32,128,0, nullptr,0,0, 512,128,128,1, GF_BIAS|GF_ACCUM);
    k_f2bf<<<(512*128+255)/256, 256, 0, stream>>>(a_f32, a_relu, (long)512*128, 1);
    gemm(stream, a_relu,128,0, Wt_b1a,128,0, PF(P_AB1L1_B), nullptr,0,0, t_abf,128,0, 512,128,128,1, GF_BIAS|GF_RELU);
    gemm(stream, t_abf, 128,0, Wt_b1b,128,0, PF(P_AB1L2_B), a_f32,128,0, nullptr,0,0, 512,128,128,1, GF_BIAS|GF_ACCUM);
    k_angle_out<<<4, 128, 0, stream>>>(a_f32, PF(P_AOUT_W), PF(P_AOUT_B), unnorm_o, ang_o);
    // torsion -> frames -> atoms (uses NEW quats/trans)
    k_frames<<<8, 64, 0, stream>>>(aatype, dframes, gidx, amask, lit, quats, trans,
                                   ang_o, side44_o, pos_o);
  }
  // final s output
  k_copy<<<(512*384+255)/256, 256, 0, stream>>>(s_cur, out + OFF_SFINAL, (long)512*384);
#undef PF
}
